// Experts_2027224564063
// MI455X (gfx1250) — matrix operand order repaired and checked by reading, not compiled, not run on hardware
//
#include <hip/hip_runtime.h>
#include <math.h>

// Problem constants (from reference): B=8, N=512, D=1024, E=8, F=2048
#define Bsz 8
#define Nsz 512
#define Dsz 1024
#define Esz 8
#define Fsz 2048
#define Tsz (Bsz * Nsz)   // 4096 tokens

// Tiling: 128x128 block tile, K-step 32, 8 waves (wave32) per block,
// each wave owns a 32(M) x 64(N) sub-tile = 2 A-frags x 4 B-frags = 8 WMMA/K-step.
#define TM 128
#define TN 128
#define TK 32
#define PITCH 40          // bf16 elements per LDS row (80B, 16B-aligned, bank-staggered)

typedef __attribute__((ext_vector_type(16))) __bf16 v16bf;
typedef __attribute__((ext_vector_type(8)))  float  v8f;

union Frag { v16bf v; uint4 q[2]; };

__device__ __forceinline__ float gelu_exact(float v) {
    // approximate=False gelu: 0.5*x*(1+erf(x/sqrt(2)))
    return 0.5f * v * (1.0f + erff(v * 0.70710678118654752440f));
}

// ---------------------------------------------------------------------------
// Kernel 1: top-1 routing. Build per-expert compacted slot lists.
// ---------------------------------------------------------------------------
__global__ __launch_bounds__(1024) void moe_route_kernel(
    const float* __restrict__ disp, const float* __restrict__ comb,
    int* __restrict__ offs, int* __restrict__ cnt,
    int* __restrict__ idx, float* __restrict__ gate)
{
    __shared__ int lcnt[Esz];
    __shared__ int loff[Esz];
    __shared__ int lcur[Esz];
    const int tid = threadIdx.x;
    if (tid < Esz) lcnt[tid] = 0;
    __syncthreads();

    for (int t = tid; t < Tsz; t += blockDim.x) {
        const float* dr = disp + (size_t)t * Esz;
        int e = 0; float best = dr[0];
        #pragma unroll
        for (int j = 1; j < Esz; ++j) { float v = dr[j]; if (v > best) { best = v; e = j; } }
        atomicAdd(&lcnt[e], 1);
    }
    __syncthreads();
    if (tid == 0) {
        int s = 0;
        for (int j = 0; j < Esz; ++j) { loff[j] = s; lcur[j] = s; s += lcnt[j]; }
    }
    __syncthreads();
    if (tid < Esz) { offs[tid] = loff[tid]; cnt[tid] = lcnt[tid]; }

    for (int t = tid; t < Tsz; t += blockDim.x) {
        const float* dr = disp + (size_t)t * Esz;
        int e = 0; float best = dr[0];
        #pragma unroll
        for (int j = 1; j < Esz; ++j) { float v = dr[j]; if (v > best) { best = v; e = j; } }
        const int slot = atomicAdd(&lcur[e], 1);
        idx[slot]  = t;
        gate[slot] = comb[(size_t)t * Esz + e];
    }
}

// ---------------------------------------------------------------------------
// Kernel 2: H[slot,:] = gelu( x[idx[slot],:] @ w1[e] + b1[e] )   (bf16 WMMA)
// ---------------------------------------------------------------------------
__global__ __launch_bounds__(256) void moe_gemm1_kernel(
    const float* __restrict__ x, const float* __restrict__ w1,
    const float* __restrict__ b1,
    const int* __restrict__ idx, const int* __restrict__ offs,
    const int* __restrict__ cnt, __bf16* __restrict__ H)
{
    const int e     = blockIdx.z;
    const int count = cnt[e];
    const int mt    = blockIdx.y;
    if (mt * TM >= count) return;
    const int base = offs[e];
    const int n0   = blockIdx.x * TN;

    __shared__ __align__(16) __bf16 Asm[TM][PITCH];   // [m][k], K contiguous
    __shared__ __align__(16) __bf16 Bsm[TN][PITCH];   // [n][k], K contiguous (transposed w1)

    const int tid  = threadIdx.x;
    const int lane = tid & 31;
    const int wid  = tid >> 5;          // 8 waves
    const int wm   = (wid & 3) * 32;    // wave M offset (2 frags)
    const int wn   = (wid >> 2) * 64;   // wave N offset (4 frags)
    const int nlo  = lane & 15;
    const int hi   = lane >> 4;

    // A-tile loader: 128 rows x 32 K fp32, 16 floats/thread
    const int ar    = tid >> 1;         // 0..127
    const int akc   = (tid & 1) * 16;   // 0 or 16
    const int aslot = mt * TM + ar;
    const int atok  = (aslot < count) ? idx[base + aslot] : -1;
    const float* arow = (atok >= 0) ? (x + (size_t)atok * Dsz) : nullptr;

    // B-tile loader: 32 K rows x 128 N fp32, 16 floats/thread, transpose into LDS
    const int bk = tid >> 3;            // 0..31
    const int bn = (tid & 7) * 16;      // 0..112
    const float* wp = w1 + (size_t)e * Dsz * Fsz + (size_t)bk * Fsz + n0 + bn;

    v8f acc[2][4];
    #pragma unroll
    for (int mi = 0; mi < 2; ++mi)
        #pragma unroll
        for (int ni = 0; ni < 4; ++ni) acc[mi][ni] = (v8f){};

    for (int k0 = 0; k0 < Dsz; k0 += TK) {
        // ---- stage A (gathered x row, fp32 -> bf16, 2x16B vector LDS stores) ----
        {
            __bf16 tmp[16];
            #pragma unroll
            for (int q = 0; q < 4; ++q) {
                float4 f = make_float4(0.f, 0.f, 0.f, 0.f);
                if (arow) f = *reinterpret_cast<const float4*>(arow + k0 + akc + q * 4);
                tmp[q*4+0] = (__bf16)f.x; tmp[q*4+1] = (__bf16)f.y;
                tmp[q*4+2] = (__bf16)f.z; tmp[q*4+3] = (__bf16)f.w;
            }
            *reinterpret_cast<uint4*>(&Asm[ar][akc])     = reinterpret_cast<const uint4*>(tmp)[0];
            *reinterpret_cast<uint4*>(&Asm[ar][akc + 8]) = reinterpret_cast<const uint4*>(tmp)[1];
        }
        // ---- stage B (w1 slab, fp32 -> bf16, transpose [k][n] -> [n][k]) ----
        {
            const float* p = wp + (size_t)k0 * Fsz;
            #pragma unroll
            for (int q = 0; q < 4; ++q) {
                float4 g = *reinterpret_cast<const float4*>(p + q * 4);
                Bsm[bn + q*4 + 0][bk] = (__bf16)g.x;
                Bsm[bn + q*4 + 1][bk] = (__bf16)g.y;
                Bsm[bn + q*4 + 2][bk] = (__bf16)g.z;
                Bsm[bn + q*4 + 3][bk] = (__bf16)g.w;
            }
            if (k0 + TK < Dsz) {
                __builtin_prefetch(p + (size_t)TK * Fsz, 0, 0);       // next w1 slab
                if (arow) __builtin_prefetch(arow + k0 + TK + akc, 0, 0);
            }
        }
        __syncthreads();

        // ---- fragments per ISA VGPR layout ----
        Frag a[2], b[4];
        #pragma unroll
        for (int mi = 0; mi < 2; ++mi) {
            const int r = wm + mi * 16 + nlo;
            a[mi].q[0] = *reinterpret_cast<const uint4*>(&Asm[r][hi * 8]);
            a[mi].q[1] = *reinterpret_cast<const uint4*>(&Asm[r][hi * 8 + 16]);
        }
        #pragma unroll
        for (int ni = 0; ni < 4; ++ni) {
            const int c = wn + ni * 16 + nlo;
            b[ni].q[0] = *reinterpret_cast<const uint4*>(&Bsm[c][hi * 8]);
            b[ni].q[1] = *reinterpret_cast<const uint4*>(&Bsm[c][hi * 8 + 16]);
        }
        #pragma unroll
        for (int mi = 0; mi < 2; ++mi)
            #pragma unroll
            for (int ni = 0; ni < 4; ++ni)
                acc[mi][ni] = __builtin_amdgcn_wmma_f32_16x16x32_bf16(
                    false, a[mi].v, false, b[ni].v, (short)0, acc[mi][ni], false, false);

        __syncthreads();
    }

    // ---- epilogue: bias + exact gelu, store bf16 H ----
    #pragma unroll
    for (int mi = 0; mi < 2; ++mi) {
        #pragma unroll
        for (int i = 0; i < 8; ++i) {
            const int ml   = wm + mi * 16 + hi * 8 + i;   // C layout: M=i / 8+i per half
            const int slot = mt * TM + ml;
            if (slot < count) {
                const size_t hrow = (size_t)(base + slot) * Fsz;
                #pragma unroll
                for (int ni = 0; ni < 4; ++ni) {
                    const int nc = n0 + wn + ni * 16 + nlo;
                    const float v = acc[mi][ni][i] + b1[e * Fsz + nc];
                    H[hrow + nc] = (__bf16)gelu_exact(v);
                }
            }
        }
    }
}

// ---------------------------------------------------------------------------
// Kernel 3: y[idx[slot],:] = gate[slot] * (H[slot,:] @ w2[e]) + b2
// A-tile staged with GLOBAL_LOAD_ASYNC_TO_LDS_B128 (pure bf16 copy, ASYNCcnt).
// ---------------------------------------------------------------------------
__global__ __launch_bounds__(256) void moe_gemm2_kernel(
    const __bf16* __restrict__ H, const float* __restrict__ w2,
    const float* __restrict__ b2, const float* __restrict__ gate,
    const int* __restrict__ idx, const int* __restrict__ offs,
    const int* __restrict__ cnt, float* __restrict__ y)
{
    const int e     = blockIdx.z;
    const int count = cnt[e];
    const int mt    = blockIdx.y;
    if (mt * TM >= count) return;           // count >= 1 past this point
    const int base = offs[e];
    const int n0   = blockIdx.x * TN;

    __shared__ __align__(16) __bf16 Asm[TM][PITCH];
    __shared__ __align__(16) __bf16 Bsm[TN][PITCH];

    const int tid  = threadIdx.x;
    const int lane = tid & 31;
    const int wid  = tid >> 5;
    const int wm   = (wid & 3) * 32;
    const int wn   = (wid >> 2) * 64;
    const int nlo  = lane & 15;
    const int hi   = lane >> 4;

    // A-tile loader: 128 rows x 32 bf16; clamp tail rows to a valid slot so the
    // async copy is unconditional (garbage rows never stored: epilogue guards).
    const int ar     = tid >> 1;
    const int akc    = (tid & 1) * 16;
    const int aslot  = mt * TM + ar;
    const int aclamp = (aslot < count) ? aslot : (count - 1);
    // Per-lane 32-bit byte offset into H (max ~16.8MB, fits signed 32-bit).
    const unsigned arowoff = (unsigned)((size_t)(base + aclamp) * Fsz * 2) + (unsigned)(akc * 2);
    // Wave-relative LDS byte address (low 32 bits of generic shared address).
    const unsigned lds0 = (unsigned)(size_t)&Asm[ar][akc];

    const int bk = tid >> 3;
    const int bn = (tid & 7) * 16;
    const float* wp = w2 + (size_t)e * Fsz * Dsz + (size_t)bk * Dsz + n0 + bn;

    v8f acc[2][4];
    #pragma unroll
    for (int mi = 0; mi < 2; ++mi)
        #pragma unroll
        for (int ni = 0; ni < 4; ++ni) acc[mi][ni] = (v8f){};

    for (int k0 = 0; k0 < Fsz; k0 += TK) {
        // ---- stage A: async DMA H -> LDS (2 x 16B per thread, no VGPR round-trip)
        {
            const unsigned go = arowoff + (unsigned)(k0 * 2);
            asm volatile("global_load_async_to_lds_b128 %0, %1, %2"
                         :: "v"(lds0), "v"(go), "s"((const void*)H) : "memory");
            asm volatile("global_load_async_to_lds_b128 %0, %1, %2"
                         :: "v"(lds0 + 16u), "v"(go + 16u), "s"((const void*)H) : "memory");
        }
        // ---- stage B (w2 slab, fp32 -> bf16, transpose) ----
        {
            const float* p = wp + (size_t)k0 * Dsz;
            #pragma unroll
            for (int q = 0; q < 4; ++q) {
                float4 g = *reinterpret_cast<const float4*>(p + q * 4);
                Bsm[bn + q*4 + 0][bk] = (__bf16)g.x;
                Bsm[bn + q*4 + 1][bk] = (__bf16)g.y;
                Bsm[bn + q*4 + 2][bk] = (__bf16)g.z;
                Bsm[bn + q*4 + 3][bk] = (__bf16)g.w;
            }
            if (k0 + TK < Fsz) __builtin_prefetch(p + (size_t)TK * Dsz, 0, 0);
        }
        // Own wave's async copies must land in LDS before the barrier.
        asm volatile("s_wait_asynccnt 0" ::: "memory");
        __syncthreads();

        Frag a[2], b[4];
        #pragma unroll
        for (int mi = 0; mi < 2; ++mi) {
            const int r = wm + mi * 16 + nlo;
            a[mi].q[0] = *reinterpret_cast<const uint4*>(&Asm[r][hi * 8]);
            a[mi].q[1] = *reinterpret_cast<const uint4*>(&Asm[r][hi * 8 + 16]);
        }
        #pragma unroll
        for (int ni = 0; ni < 4; ++ni) {
            const int c = wn + ni * 16 + nlo;
            b[ni].q[0] = *reinterpret_cast<const uint4*>(&Bsm[c][hi * 8]);
            b[ni].q[1] = *reinterpret_cast<const uint4*>(&Bsm[c][hi * 8 + 16]);
        }
        #pragma unroll
        for (int mi = 0; mi < 2; ++mi)
            #pragma unroll
            for (int ni = 0; ni < 4; ++ni)
                acc[mi][ni] = __builtin_amdgcn_wmma_f32_16x16x32_bf16(
                    false, a[mi].v, false, b[ni].v, (short)0, acc[mi][ni], false, false);

        __syncthreads();
    }

    // ---- epilogue: gate * acc + b2, scatter to token row (each token written once)
    #pragma unroll
    for (int mi = 0; mi < 2; ++mi) {
        #pragma unroll
        for (int i = 0; i < 8; ++i) {
            const int ml   = wm + mi * 16 + hi * 8 + i;
            const int slot = mt * TM + ml;
            if (slot < count) {
                const int gs  = base + slot;
                const int tok = idx[gs];
                const float gw = gate[gs];
                #pragma unroll
                for (int ni = 0; ni < 4; ++ni) {
                    const int nc = n0 + wn + ni * 16 + nlo;
                    y[(size_t)tok * Dsz + nc] = gw * acc[mi][ni][i] + b2[nc];
                }
            }
        }
    }
}

// ---------------------------------------------------------------------------
extern "C" void kernel_launch(void* const* d_in, const int* in_sizes, int n_in,
                              void* d_out, int out_size, void* d_ws, size_t ws_size,
                              hipStream_t stream) {
    const float* x    = (const float*)d_in[0];
    const float* disp = (const float*)d_in[1];
    const float* comb = (const float*)d_in[2];
    const float* w1   = (const float*)d_in[3];
    const float* b1   = (const float*)d_in[4];
    const float* w2   = (const float*)d_in[5];
    const float* b2   = (const float*)d_in[6];
    float* y = (float*)d_out;

    // Workspace layout
    char*   ws   = (char*)d_ws;
    int*    offs = (int*)(ws);                  //  8 ints
    int*    cnt  = (int*)(ws + 32);             //  8 ints
    int*    idx  = (int*)(ws + 64);             //  T ints
    float*  gate = (float*)(ws + 64 + Tsz * 4); //  T floats
    __bf16* H    = (__bf16*)(ws + 65536);       //  T*F bf16 = 16 MB

    moe_route_kernel<<<1, 1024, 0, stream>>>(disp, comb, offs, cnt, idx, gate);

    dim3 g1(Fsz / TN, Tsz / TM, Esz);   // (16, 32, 8) — early-exit past count[e]
    moe_gemm1_kernel<<<g1, 256, 0, stream>>>(x, w1, b1, idx, offs, cnt, H);

    dim3 g2(Dsz / TN, Tsz / TM, Esz);   // (8, 32, 8)
    moe_gemm2_kernel<<<g2, 256, 0, stream>>>(H, w2, b2, gate, idx, offs, cnt, y);
}
